// ExplainableLSTM_82394652606944
// MI455X (gfx1250) — matrix operand order repaired and checked by reading, not compiled, not run on hardware
//
#include <hip/hip_runtime.h>

#define B_   128
#define S_   256
#define F_   4
#define V_   1024
#define E_   512
#define H_   1024

typedef __bf16 bf16;
typedef __attribute__((ext_vector_type(8)))  bf16  v8bf;
typedef __attribute__((ext_vector_type(16))) bf16  v16bf;
typedef __attribute__((ext_vector_type(8)))  float v8f;

// ---------------------------------------------------------------------------
// WMMA fragment loaders, contiguous k order (k = 16*(lane/16) + s) shared by A and B
// A: 16x32 bf16, row-major source, lane<16 holds K 0-15, lane>=16 holds
//    K 16-31, M = lane&15.
// B: 32x16 bf16 where B[k,n] = W[n,k] (weights stored [N,K] row-major).
//    lane holds column n = lane&15; lane<16 -> K 0-15, lane>=16 -> K 16-31.
// ---------------------------------------------------------------------------
__device__ inline v16bf load_a_frag(const bf16* __restrict__ A, int lda,
                                    int m0, int k0, int lane) {
  int m  = m0 + (lane & 15);
  int kb = k0 + ((lane < 16) ? 0 : 16);
  const bf16* row = A + (long)m * lda + kb;
  v8bf lo = *(const v8bf*)(row);
  v8bf hi = *(const v8bf*)(row + 8);
  return __builtin_shufflevector(lo, hi, 0, 1, 2, 3, 4, 5, 6, 7,
                                         8, 9, 10, 11, 12, 13, 14, 15);
}

__device__ inline v16bf load_b_frag(const bf16* __restrict__ W, int ldw,
                                    int n0, int k0, int lane) {
  int n  = n0 + (lane & 15);
  int kb = k0 + ((lane < 16) ? 0 : 16);
  const bf16* p = W + (long)n * ldw + kb;
  __builtin_prefetch(p + 32, 0, 1);
  return *(const v16bf*)p;
}

__device__ inline float sigmoidf_(float x) { return 1.0f / (1.0f + __expf(-x)); }

// ---------------------------------------------------------------------------
// Elementwise helpers
// ---------------------------------------------------------------------------
__global__ __launch_bounds__(256) void f32_to_bf16_kernel(
    const float* __restrict__ in, bf16* __restrict__ out, long n) {
  for (long i = (long)blockIdx.x * blockDim.x + threadIdx.x; i < n;
       i += (long)gridDim.x * blockDim.x)
    out[i] = (bf16)in[i];
}

__global__ __launch_bounds__(256) void add_bias_kernel(
    const float* __restrict__ a, const float* __restrict__ b,
    float* __restrict__ out, int n) {
  int i = blockIdx.x * blockDim.x + threadIdx.x;
  if (i < n) out[i] = a[i] + b[i];
}

__global__ __launch_bounds__(256) void zero_u32_kernel(
    unsigned int* __restrict__ p, long n) {
  for (long i = (long)blockIdx.x * blockDim.x + threadIdx.x; i < n;
       i += (long)gridDim.x * blockDim.x)
    p[i] = 0u;
}

// ---------------------------------------------------------------------------
// Embedding gather-sum:  out[b*S+s, :] = sum_f emb[f, x[b,s,f], :]  (bf16 out)
// ---------------------------------------------------------------------------
__global__ __launch_bounds__(256) void embed_sum_kernel(
    const int* __restrict__ x, const float* __restrict__ emb,
    bf16* __restrict__ out) {
  const long row = blockIdx.x;  // b*S + s (matches x's [B,S,F] layout)
  const int i0 = x[row * F_ + 0];
  const int i1 = x[row * F_ + 1];
  const int i2 = x[row * F_ + 2];
  const int i3 = x[row * F_ + 3];
  const float* t0 = emb + ((long)0 * V_ + i0) * E_;
  const float* t1 = emb + ((long)1 * V_ + i1) * E_;
  const float* t2 = emb + ((long)2 * V_ + i2) * E_;
  const float* t3 = emb + ((long)3 * V_ + i3) * E_;
  for (int e = threadIdx.x; e < E_; e += 256) {
    float s = t0[e] + t1[e] + t2[e] + t3[e];
    out[row * E_ + e] = (bf16)s;
  }
}

// ---------------------------------------------------------------------------
// Projection GEMM:  x_in = emb_sum @ proj_w.T + proj_b   (one-shot, 17 GFLOP)
//   Output written TIME-MAJOR bf16: x_in[(s*B+b)*E + n].
// ---------------------------------------------------------------------------
__global__ __launch_bounds__(256) void proj_gemm_kernel(
    const bf16* __restrict__ A, const bf16* __restrict__ W,
    const float* __restrict__ bias, bf16* __restrict__ out) {
  const int lane = threadIdx.x & 31;
  const int wave = threadIdx.x >> 5;
  const int m0 = blockIdx.x * 128 + wave * 16;
  const int n0base = blockIdx.y * 64;

  v8f acc[4] = {};
  for (int k0 = 0; k0 < E_; k0 += 32) {
    v16bf a = load_a_frag(A, E_, m0, k0, lane);
#pragma unroll
    for (int j = 0; j < 4; ++j) {
      v16bf b = load_b_frag(W, E_, n0base + j * 16, k0, lane);
      acc[j] = __builtin_amdgcn_wmma_f32_16x16x32_bf16(
          false, a, false, b, (short)0, acc[j], false, false);
    }
  }
#pragma unroll
  for (int j = 0; j < 4; ++j) {
    const int n = n0base + j * 16 + (lane & 15);
    const float bv = bias[n];
#pragma unroll
    for (int r = 0; r < 8; ++r) {
      int m = m0 + r + ((lane < 16) ? 0 : 8);
      int b = m / S_;
      int s = m % S_;
      out[((long)s * B_ + b) * E_ + n] = (bf16)(acc[j][r] + bv);
    }
  }
}

// ---------------------------------------------------------------------------
// Fused LSTM layer step, async-DMA weight staging, 4-deep LDS ring:
//   gates[B,4H] = xin[B,KIN] @ Wih.T + hprev[B,H] @ Whh.T + bias
//   Block owns 16 h-columns -> 4 gate tiles (cols n, H+n, 2H+n, 3H+n);
//   8 waves cover M=128.  Per 32-K chunk, the 4-KB weight tile is DMA'd to
//   LDS once per block (global_load_async_to_lds_b128, ASYNCcnt-tracked,
//   lookahead 2, ring of 4 buffers -> one barrier per chunk), while the
//   per-wave A fragment is software-pipelined in registers.
// ---------------------------------------------------------------------------
template <int KIN>
__global__ __launch_bounds__(256) void lstm_step_kernel(
    const bf16* __restrict__ xin,     // [B, KIN]
    const bf16* __restrict__ hprev,   // [B, H]
    const bf16* __restrict__ Wih,     // [4H, KIN]
    const bf16* __restrict__ Whh,     // [4H, H]
    const float* __restrict__ bias,   // [4H] (b_ih + b_hh)
    float* __restrict__ cstate,       // [B, H] in/out
    float* __restrict__ hfout,        // optional fp32 h out (strided rows)
    long hf_stride,
    bf16* __restrict__ hbout)         // [B, H] bf16 h out (next-step input)
{
  // stage[buf][gate][n (16)][k (32)] bf16 : 4 x 4KB ring
  __shared__ __align__(16) bf16 stage[4][4 * 16 * 32];

  const int tid  = threadIdx.x;
  const int lane = tid & 31;
  const int wave = tid >> 5;
  const int m0 = wave * 16;            // batch tile (8 waves * 16 = 128 = B)
  const int n0 = blockIdx.x * 16;      // h-column tile

  // Per-thread async-copy slot: one 16B (8 x bf16) transfer per chunk.
  const int sg   = tid >> 6;           // gate 0..3
  const int sidx = tid & 63;
  const int sn   = sidx >> 2;          // weight row within tile, 0..15
  const int sk   = (sidx & 3) * 8;     // K sub-offset: 0,8,16,24

  constexpr int nchunks = (KIN + H_) / 32;   // 48 (layer0) or 64 (layer1)

  auto issue = [&](int c) {
    const int k0 = c * 32;
    const bf16* src;
    if (k0 < KIN)
      src = Wih + (long)(sg * H_ + n0 + sn) * KIN + (k0 + sk);
    else
      src = Whh + (long)(sg * H_ + n0 + sn) * H_ + (k0 - KIN + sk);
    unsigned lds = (unsigned)(uintptr_t)&stage[c & 3][(sg * 16 + sn) * 32 + sk];
    unsigned long long ga = (unsigned long long)(uintptr_t)src;
    asm volatile("global_load_async_to_lds_b128 %0, %1, off"
                 :: "v"(lds), "v"(ga) : "memory");
  };
  auto loadA = [&](int c) -> v16bf {
    const int k0 = c * 32;
    if (k0 < KIN) return load_a_frag(xin, KIN, m0, k0, lane);
    return load_a_frag(hprev, H_, m0, k0 - KIN, lane);
  };

  // Accumulators initialized with the (broadcast) bias.
  v8f acc[4];
  {
    const int n = n0 + (lane & 15);
#pragma unroll
    for (int g = 0; g < 4; ++g) {
      const float bv = bias[g * H_ + n];
#pragma unroll
      for (int r = 0; r < 8; ++r) acc[g][r] = bv;
    }
  }

  // Prologue: two chunks in flight, first A fragment in registers.
  issue(0);
  issue(1);
  v16bf a_cur = loadA(0);

#pragma unroll 4
  for (int c = 0; c < nchunks; ++c) {
    // Async completion is in-order: asynccnt<=1 guarantees chunk c landed
    // (only c+1 may still be in flight).  Last chunk: wait for all.
    if (c <= nchunks - 2) asm volatile("s_wait_asynccnt 0x1" ::: "memory");
    else                  asm volatile("s_wait_asynccnt 0x0" ::: "memory");
    __syncthreads();  // chunk c visible to all waves; also orders the reads
                      // of buffer (c+2)&3 (done at iter c-2) before re-issue

    if (c + 2 < nchunks) issue(c + 2);

    v16bf a_next;
    if (c + 1 < nchunks) a_next = loadA(c + 1);  // overlaps with WMMAs below

    const bf16* sb = &stage[c & 3][0];
#pragma unroll
    for (int g = 0; g < 4; ++g) {
      const bf16* p = sb + (g * 16 + (lane & 15)) * 32 + ((lane < 16) ? 0 : 16);
      v8bf blo = *(const v8bf*)(p);
      v8bf bhi = *(const v8bf*)(p + 8);
      v16bf b = __builtin_shufflevector(blo, bhi, 0, 1, 2, 3, 4, 5, 6, 7,
                                                  8, 9, 10, 11, 12, 13, 14, 15);
      acc[g] = __builtin_amdgcn_wmma_f32_16x16x32_bf16(
          false, a_cur, false, b, (short)0, acc[g], false, false);
    }
    a_cur = a_next;
  }

  // Fused gate nonlinearities + cell/hidden update (bias already in acc).
  const int n = n0 + (lane & 15);
#pragma unroll
  for (int r = 0; r < 8; ++r) {
    const int m = m0 + r + ((lane < 16) ? 0 : 8);
    const float ig = sigmoidf_(acc[0][r]);
    const float fg = sigmoidf_(acc[1][r]);
    const float gg = tanhf(acc[2][r]);
    const float og = sigmoidf_(acc[3][r]);
    const long ci = (long)m * H_ + n;
    const float cn = fg * cstate[ci] + ig * gg;
    cstate[ci] = cn;
    const float hn = og * tanhf(cn);
    if (hfout) hfout[(long)m * hf_stride + n] = hn;
    hbout[ci] = (bf16)hn;
  }
}

// ---------------------------------------------------------------------------
// Host-side orchestration
// ---------------------------------------------------------------------------
extern "C" void kernel_launch(void* const* d_in, const int* in_sizes, int n_in,
                              void* d_out, int out_size, void* d_ws,
                              size_t ws_size, hipStream_t stream) {
  const int*   x      = (const int*)  d_in[0];
  const float* emb    = (const float*)d_in[1];
  const float* proj_w = (const float*)d_in[2];
  const float* proj_b = (const float*)d_in[3];
  const float* W_ih0  = (const float*)d_in[4];
  const float* W_hh0  = (const float*)d_in[5];
  const float* b_ih0  = (const float*)d_in[6];
  const float* b_hh0  = (const float*)d_in[7];
  const float* W_ih1  = (const float*)d_in[8];
  const float* W_hh1  = (const float*)d_in[9];
  const float* b_ih1  = (const float*)d_in[10];
  const float* b_hh1  = (const float*)d_in[11];
  float* out = (float*)d_out;

  // ---- workspace carve-up (256B aligned) ----
  char* ws = (char*)d_ws;
  size_t off = 0;
  auto alloc = [&](size_t bytes) -> char* {
    off = (off + 255) & ~(size_t)255;
    char* p = ws + off;
    off += bytes;
    return p;
  };
  const long M = (long)B_ * S_;
  bf16*  emb_sum = (bf16*) alloc(M * E_ * sizeof(bf16));       // [B*S, E]
  bf16*  x_in    = (bf16*) alloc(M * E_ * sizeof(bf16));       // [S, B, E]
  bf16*  pw      = (bf16*) alloc((long)E_ * E_ * sizeof(bf16));
  bf16*  wih0    = (bf16*) alloc((long)4 * H_ * E_ * sizeof(bf16));
  bf16*  whh0    = (bf16*) alloc((long)4 * H_ * H_ * sizeof(bf16));
  bf16*  wih1    = (bf16*) alloc((long)4 * H_ * H_ * sizeof(bf16));
  bf16*  whh1    = (bf16*) alloc((long)4 * H_ * H_ * sizeof(bf16));
  float* bias0   = (float*)alloc((long)4 * H_ * sizeof(float));
  float* bias1   = (float*)alloc((long)4 * H_ * sizeof(float));
  float* c0      = (float*)alloc((long)B_ * H_ * sizeof(float));
  float* c1      = (float*)alloc((long)B_ * H_ * sizeof(float));
  bf16*  h0_a    = (bf16*) alloc((long)B_ * H_ * sizeof(bf16));
  bf16*  h0_b    = (bf16*) alloc((long)B_ * H_ * sizeof(bf16));
  bf16*  h1_a    = (bf16*) alloc((long)B_ * H_ * sizeof(bf16));
  bf16*  h1_b    = (bf16*) alloc((long)B_ * H_ * sizeof(bf16));
  (void)ws_size; (void)in_sizes; (void)n_in; (void)out_size;

  // ---- zero initial states (c0, c1, h0_a, h1_a) ----
  zero_u32_kernel<<<512, 256, 0, stream>>>((unsigned int*)c0, (long)B_ * H_);
  zero_u32_kernel<<<512, 256, 0, stream>>>((unsigned int*)c1, (long)B_ * H_);
  zero_u32_kernel<<<256, 256, 0, stream>>>((unsigned int*)h0_a, (long)B_ * H_ / 2);
  zero_u32_kernel<<<256, 256, 0, stream>>>((unsigned int*)h1_a, (long)B_ * H_ / 2);

  // ---- weights -> bf16 (one-time; stay L2-resident through the scan) ----
  f32_to_bf16_kernel<<<1024, 256, 0, stream>>>(proj_w, pw,   (long)E_ * E_);
  f32_to_bf16_kernel<<<1024, 256, 0, stream>>>(W_ih0,  wih0, (long)4 * H_ * E_);
  f32_to_bf16_kernel<<<1024, 256, 0, stream>>>(W_hh0,  whh0, (long)4 * H_ * H_);
  f32_to_bf16_kernel<<<1024, 256, 0, stream>>>(W_ih1,  wih1, (long)4 * H_ * H_);
  f32_to_bf16_kernel<<<1024, 256, 0, stream>>>(W_hh1,  whh1, (long)4 * H_ * H_);
  add_bias_kernel<<<(4 * H_ + 255) / 256, 256, 0, stream>>>(b_ih0, b_hh0, bias0, 4 * H_);
  add_bias_kernel<<<(4 * H_ + 255) / 256, 256, 0, stream>>>(b_ih1, b_hh1, bias1, 4 * H_);

  // ---- embedding gather-sum + projection GEMM (time-major bf16 x_in) ----
  embed_sum_kernel<<<(unsigned)M, 256, 0, stream>>>(x, emb, emb_sum);
  dim3 pgrid((unsigned)(M / 128), (unsigned)(E_ / 64));
  proj_gemm_kernel<<<pgrid, 256, 0, stream>>>(emb_sum, pw, proj_b, x_in);

  // ---- LSTM scan: 2 fused WMMA layer-step kernels per timestep ----
  bf16* h0_in = h0_a;  bf16* h0_out = h0_b;
  bf16* h1_in = h1_a;  bf16* h1_out = h1_b;
  dim3 lgrid(H_ / 16);  // 64 blocks, each owns 16 h-columns across all 4 gates
  for (int t = 0; t < S_; ++t) {
    const bf16* xin_t = x_in + (long)t * B_ * E_;
    lstm_step_kernel<E_><<<lgrid, 256, 0, stream>>>(
        xin_t, h0_in, wih0, whh0, bias0, c0,
        (float*)nullptr, 0L, h0_out);
    lstm_step_kernel<H_><<<lgrid, 256, 0, stream>>>(
        h0_out, h1_in, wih1, whh1, bias1, c1,
        out + (long)t * H_, (long)S_ * H_, h1_out);
    bf16* t0 = h0_in; h0_in = h0_out; h0_out = t0;
    bf16* t1 = h1_in; h1_in = h1_out; h1_out = t1;
  }
}
